// PatchCore_60266981098005
// MI455X (gfx1250) — matrix operand order repaired and checked by reading, not compiled, not run on hardware
//
#include <hip/hip_runtime.h>

// ---------- types ----------
typedef __attribute__((ext_vector_type(16))) __bf16        v16bf;
typedef __attribute__((ext_vector_type(2)))  __bf16        v2bf;
typedef __attribute__((ext_vector_type(8)))  float         v8f;
typedef __attribute__((ext_vector_type(4)))  float         f32x4;
typedef __attribute__((ext_vector_type(4)))  unsigned int  u32x4;
typedef __attribute__((ext_vector_type(4)))  int           i32x4;
typedef __attribute__((ext_vector_type(8)))  int           i32x8;

union FragU { u32x4 u[2]; v16bf v; };

#define DIMK    1536
#define TQ      128     // query rows per workgroup tile
#define TM      128     // memory rows per workgroup tile
// fallback (convert-in-kernel) path
#define BK      32
#define LSTR    40      // 80B rows: 16B aligned, conflict-free
// TDM path
#define BK2     64
#define LSTR2   72      // 144B rows: TDM pad writes this stride; conflict-free

// ---------- f32 -> packed bf16 ----------
#if __has_builtin(__builtin_amdgcn_cvt_pk_bf16_f32)
__device__ __forceinline__ unsigned int pack_bf16(float a, float b) {
    union { v2bf v; unsigned int u; } x;
    x.v = __builtin_amdgcn_cvt_pk_bf16_f32(a, b);
    return x.u;
}
#else
__device__ __forceinline__ unsigned int pack_bf16(float a, float b) {
    unsigned int ua = __float_as_uint(a);
    unsigned int ub = __float_as_uint(b);
    ua += 0x7FFFu + ((ua >> 16) & 1u);              // RNE
    ub += 0x7FFFu + ((ub >> 16) & 1u);
    return (ua >> 16) | (ub & 0xFFFF0000u);
}
#endif

__device__ __forceinline__ void store_bf16x4(unsigned short* dst, f32x4 v) {
    unsigned int* p = (unsigned int*)dst;
    p[0] = pack_bf16(v.x, v.y);
    p[1] = pack_bf16(v.z, v.w);
}

// ---------- WMMA fragment loaders (stride-parametrized) ----------
// A (16x32 bf16): lane l -> row (l&15); kb=(l>>4)*8; V0-3 = K kb..kb+7, V4-7 = K kb+16..kb+23
template <int STRIDE>
__device__ __forceinline__ v16bf load_fragA(const unsigned short* tile, int rowBase, int lane) {
    int r  = rowBase + (lane & 15);
    int kb = (lane >> 4) * 8;
    const unsigned short* p = tile + r * STRIDE + kb;
    FragU x;
    x.u[0] = *(const u32x4*)(p);
    x.u[1] = *(const u32x4*)(p + 16);
    return x.v;
}
// B (32x16 bf16): lane l -> col (l&15); kb=(l>>4)*8; V0-3 = K kb..kb+7, V4-7 = K kb+16..kb+23
template <int STRIDE>
__device__ __forceinline__ v16bf load_fragB(const unsigned short* tile, int rowBase, int lane) {
    int n  = rowBase + (lane & 15);
    int kb = (lane >> 4) * 8;
    const unsigned short* p = tile + n * STRIDE + kb;
    FragU x;
    x.u[0] = *(const u32x4*)(p);
    x.u[1] = *(const u32x4*)(p + 16);
    return x.v;
}

// ---------- TDM 2D tile load: global(bf16, row-major) -> LDS (padded rows) ----------
// D# per ISA 8.3/8.4. data_size=2B; pad: every 32 DWORDs (=128B=BK2 elems) insert 4 DWORDs (16B)
// -> LDS row stride 144B = LSTR2 elems.
__device__ __forceinline__ void tdm_load_2d(unsigned long long gaddr, unsigned lds_byte,
                                            unsigned dim0_rem, unsigned dim1_rem,
                                            unsigned stride0, unsigned tile0, unsigned tile1) {
    u32x4 g0;
    g0.x = 0x1u;                                                  // count=1 (valid), no gather
    g0.y = lds_byte;                                              // lds_addr [63:32]
    g0.z = (unsigned)(gaddr & 0xFFFFFFFFu);                       // global_addr[31:0] -> [95:64]
    g0.w = (unsigned)((gaddr >> 32) & 0x01FFFFFFu) | (2u << 30);  // addr[56:32], type=2
    i32x8 g1;
    g1[0] = (int)((1u << 16)        // data_size = 2 bytes
                | (1u << 20)        // pad_enable
                | (4u << 22)        // pad_interval: 32 DWORDs
                | (3u << 25));      // pad_amount:   4 DWORDs
    g1[1] = (int)((dim0_rem & 0xFFFFu) << 16);                               // tensor_dim0[15:0] @ [63:48]
    g1[2] = (int)(((dim0_rem >> 16) & 0xFFFFu) | ((dim1_rem & 0xFFFFu) << 16));
    g1[3] = (int)(((dim1_rem >> 16) & 0xFFFFu) | (tile0 << 16));             // tile_dim0 @ [127:112]
    g1[4] = (int)(tile1 & 0xFFFFu);                                          // tile_dim1; tile_dim2=0
    g1[5] = (int)stride0;                                                    // tensor_dim0_stride[31:0]
    g1[6] = 0;
    g1[7] = 0;
    i32x4 z4 = {0, 0, 0, 0};
    i32x8 z8 = {0, 0, 0, 0, 0, 0, 0, 0};
    __builtin_amdgcn_tensor_load_to_lds(g0, g1, z4, z4, z8, 0);   // clang-23: 6-arg form
}

// ---------- shared epilogue: d2 = q2 + m2 - 2*dot, min over 64 m, atomic-min per q ----------
__device__ __forceinline__ void epilogue_minreduce(v8f (&acc)[2][4],
        const float* __restrict__ q2, const float* __restrict__ m2, float* __restrict__ minD2,
        int Qbase, int Mbase, int qg, int mg, int lane, int Q, int M) {
    const int half = lane >> 4;
    const int ln   = lane & 15;
    float m2v[4]; bool mok[4];
#pragma unroll
    for (int b = 0; b < 4; ++b) {
        int m  = Mbase + mg * 64 + b * 16 + ln;
        mok[b] = (m < M);
        m2v[b] = mok[b] ? m2[m] : 0.f;
    }
#pragma unroll
    for (int a = 0; a < 2; ++a) {
#pragma unroll
        for (int j = 0; j < 8; ++j) {
            int q = Qbase + qg * 32 + a * 16 + j + 8 * half;   // C/D layout: M = j + 8*(lane>=16)
            if (q >= Q) continue;
            float q2v  = q2[q];
            float dmin = 3.0e38f;
#pragma unroll
            for (int b = 0; b < 4; ++b) {
                float d2 = fmaxf(q2v + m2v[b] - 2.0f * acc[a][b][j], 0.f);
                if (mok[b]) dmin = fminf(dmin, d2);
            }
#pragma unroll
            for (int s = 8; s; s >>= 1) dmin = fminf(dmin, __shfl_xor(dmin, s, 32));
            if (ln == 0)
                atomicMin((int*)&minD2[q], __float_as_int(dmin));  // d2>=0: int order == float
        }
    }
}

// ---------- small kernels ----------
__global__ void init_min_kernel(float* minD2, int n) {
    int i = blockIdx.x * blockDim.x + threadIdx.x;
    if (i < n) minD2[i] = 3.0e38f;
}

__global__ void sqnorm_kernel(const float* __restrict__ X, float* __restrict__ out, int nrows) {
    int w    = (int)((blockIdx.x * blockDim.x + threadIdx.x) >> 5);
    int lane = threadIdx.x & 31;
    if (w >= nrows) return;
    const f32x4* row = (const f32x4*)(X + (size_t)w * DIMK);
    float s = 0.f;
#pragma unroll
    for (int i = 0; i < DIMK / (32 * 4); ++i) {
        f32x4 v = row[lane + i * 32];
        s += v.x * v.x + v.y * v.y + v.z * v.z + v.w * v.w;
    }
#pragma unroll
    for (int m = 16; m; m >>= 1) s += __shfl_xor(s, m, 32);
    if (lane == 0) out[w] = s;
}

// streaming f32 -> bf16 conversion (one-time pass)
__global__ void cvt_bf16_kernel(const float* __restrict__ X, unsigned short* __restrict__ Y,
                                long long n4) {
    long long i      = (long long)blockIdx.x * blockDim.x + threadIdx.x;
    long long stride = (long long)gridDim.x * blockDim.x;
    for (; i < n4; i += stride) {
        f32x4 v = ((const f32x4*)X)[i];
        unsigned int* p = (unsigned int*)(Y + i * 4);
        p[0] = pack_bf16(v.x, v.y);
        p[1] = pack_bf16(v.z, v.w);
    }
}

// ---------- TDM-staged GEMM (bf16 inputs in workspace) ----------
__global__ __launch_bounds__(256)
void nn_gemm_tdm_kernel(const unsigned short* __restrict__ qbf, const unsigned short* __restrict__ mbf,
                        const float* __restrict__ q2, const float* __restrict__ m2,
                        float* __restrict__ minD2, int Q, int M) {
    __shared__ __align__(16) unsigned short ldsQ[2][TQ * LSTR2];
    __shared__ __align__(16) unsigned short ldsM[2][TM * LSTR2];

    const int Qbase = blockIdx.x * TQ;
    const int Mbase = blockIdx.y * TM;
    const int wave  = threadIdx.x >> 5;
    const int lane  = threadIdx.x & 31;
    const int qg    = wave >> 1;   // 0..3
    const int mg    = wave & 1;    // 0..1

    v8f acc[2][4];
#pragma unroll
    for (int a = 0; a < 2; ++a)
#pragma unroll
        for (int b = 0; b < 4; ++b) {
            v8f z = {0.f, 0.f, 0.f, 0.f, 0.f, 0.f, 0.f, 0.f};
            acc[a][b] = z;
        }

    // generic LDS pointer low 32 bits == LDS byte offset (ISA flat aperture rule)
    const unsigned lq[2] = { (unsigned)(size_t)&ldsQ[0][0], (unsigned)(size_t)&ldsQ[1][0] };
    const unsigned lm[2] = { (unsigned)(size_t)&ldsM[0][0], (unsigned)(size_t)&ldsM[1][0] };

    const unsigned long long qaddr = (unsigned long long)(size_t)qbf + (unsigned long long)Qbase * DIMK * 2ull;
    const unsigned long long maddr = (unsigned long long)(size_t)mbf + (unsigned long long)Mbase * DIMK * 2ull;
    const unsigned qrem = (unsigned)(Q > Qbase ? Q - Qbase : 0);   // rows left (OOB rows read 0)
    const unsigned mrem = (unsigned)(M > Mbase ? M - Mbase : 0);

    if (wave == 0) {
        tdm_load_2d(qaddr, lq[0], DIMK, qrem, DIMK, BK2, TQ);
        tdm_load_2d(maddr, lm[0], DIMK, mrem, DIMK, BK2, TM);
        __builtin_amdgcn_s_wait_tensorcnt(0);
    }
    __syncthreads();

    int it = 0;
    for (int k0 = 0; k0 < DIMK; k0 += BK2, ++it) {
        const int cur  = it & 1;
        const int nxt  = cur ^ 1;
        const bool more = (k0 + BK2) < DIMK;

        if (more && wave == 0) {   // async DMA of next chunk overlaps WMMA below
            tdm_load_2d(qaddr + (unsigned long long)(k0 + BK2) * 2ull, lq[nxt],
                        (unsigned)(DIMK - (k0 + BK2)), qrem, DIMK, BK2, TQ);
            tdm_load_2d(maddr + (unsigned long long)(k0 + BK2) * 2ull, lm[nxt],
                        (unsigned)(DIMK - (k0 + BK2)), mrem, DIMK, BK2, TM);
        }

#pragma unroll
        for (int ks = 0; ks < BK2; ks += 32) {   // two 16x16x32 K-steps per chunk
            const unsigned short* tq = &ldsQ[cur][ks];
            const unsigned short* tm = &ldsM[cur][ks];
            v16bf a0 = load_fragA<LSTR2>(tq, qg * 32,      lane);
            v16bf a1 = load_fragA<LSTR2>(tq, qg * 32 + 16, lane);
#pragma unroll
            for (int b = 0; b < 4; ++b) {
                v16bf bb = load_fragB<LSTR2>(tm, mg * 64 + b * 16, lane);
                acc[0][b] = __builtin_amdgcn_wmma_f32_16x16x32_bf16(
                                false, a0, false, bb, (short)0, acc[0][b], false, false);
                acc[1][b] = __builtin_amdgcn_wmma_f32_16x16x32_bf16(
                                false, a1, false, bb, (short)0, acc[1][b], false, false);
            }
        }

        if (more && wave == 0) __builtin_amdgcn_s_wait_tensorcnt(0);
        __syncthreads();
    }

    epilogue_minreduce(acc, q2, m2, minD2, Qbase, Mbase, qg, mg, lane, Q, M);
}

// ---------- fallback GEMM (convert f32->bf16 in-kernel, double-buffered) ----------
__global__ __launch_bounds__(256)
void nn_gemm_kernel(const float* __restrict__ query, const float* __restrict__ memv,
                    const float* __restrict__ q2,    const float* __restrict__ m2,
                    float* __restrict__ minD2, int Q, int M) {
    __shared__ __align__(16) unsigned short ldsQ[2][TQ * LSTR];
    __shared__ __align__(16) unsigned short ldsM[2][TM * LSTR];

    const int Qbase = blockIdx.x * TQ;
    const int Mbase = blockIdx.y * TM;
    const int tid   = threadIdx.x;
    const int wave  = tid >> 5;
    const int lane  = tid & 31;
    const int qg    = wave >> 1;
    const int mg    = wave & 1;

    const float* qsrc[4]; const float* msrc[4];
    int dstoff[4]; bool qok[4], mok_st[4];
#pragma unroll
    for (int i = 0; i < 4; ++i) {
        int f   = tid + i * 256;
        int row = f >> 3;
        int c4  = (f & 7) << 2;
        dstoff[i] = row * LSTR + c4;
        int qr = Qbase + row, mr = Mbase + row;
        qok[i]    = qr < Q;
        mok_st[i] = mr < M;
        qsrc[i] = query + (size_t)(qok[i]    ? qr : 0) * DIMK + c4;
        msrc[i] = memv  + (size_t)(mok_st[i] ? mr : 0) * DIMK + c4;
    }

    v8f acc[2][4];
#pragma unroll
    for (int a = 0; a < 2; ++a)
#pragma unroll
        for (int b = 0; b < 4; ++b) {
            v8f z = {0.f, 0.f, 0.f, 0.f, 0.f, 0.f, 0.f, 0.f};
            acc[a][b] = z;
        }

    const f32x4 fz = {0.f, 0.f, 0.f, 0.f};
    f32x4 rq[4], rm[4];
#pragma unroll
    for (int i = 0; i < 4; ++i) {
        rq[i] = qok[i]    ? *(const f32x4*)(qsrc[i]) : fz;
        rm[i] = mok_st[i] ? *(const f32x4*)(msrc[i]) : fz;
    }
#pragma unroll
    for (int i = 0; i < 4; ++i) {
        store_bf16x4(&ldsQ[0][dstoff[i]], rq[i]);
        store_bf16x4(&ldsM[0][dstoff[i]], rm[i]);
    }
    __syncthreads();

    int it = 0;
    for (int k0 = 0; k0 < DIMK; k0 += BK, ++it) {
        const int cur  = it & 1;
        const int nxt  = cur ^ 1;
        const bool more = (k0 + BK) < DIMK;
        if (more) {
#pragma unroll
            for (int i = 0; i < 4; ++i) {
                rq[i] = qok[i]    ? *(const f32x4*)(qsrc[i] + k0 + BK) : fz;
                rm[i] = mok_st[i] ? *(const f32x4*)(msrc[i] + k0 + BK) : fz;
            }
            if (mok_st[0] && k0 + 2 * BK < DIMK)
                __builtin_prefetch(msrc[0] + k0 + 2 * BK, 0, 0);
        }
        v16bf a0 = load_fragA<LSTR>(ldsQ[cur], qg * 32,      lane);
        v16bf a1 = load_fragA<LSTR>(ldsQ[cur], qg * 32 + 16, lane);
#pragma unroll
        for (int b = 0; b < 4; ++b) {
            v16bf bb = load_fragB<LSTR>(ldsM[cur], mg * 64 + b * 16, lane);
            acc[0][b] = __builtin_amdgcn_wmma_f32_16x16x32_bf16(
                            false, a0, false, bb, (short)0, acc[0][b], false, false);
            acc[1][b] = __builtin_amdgcn_wmma_f32_16x16x32_bf16(
                            false, a1, false, bb, (short)0, acc[1][b], false, false);
        }
        if (more) {
#pragma unroll
            for (int i = 0; i < 4; ++i) {
                store_bf16x4(&ldsQ[nxt][dstoff[i]], rq[i]);
                store_bf16x4(&ldsM[nxt][dstoff[i]], rm[i]);
            }
        }
        __syncthreads();
    }

    epilogue_minreduce(acc, q2, m2, minD2, Qbase, Mbase, qg, mg, lane, Q, M);
}

__global__ void finalize_kernel(const float* __restrict__ minD2, float* __restrict__ out, int Q) {
    __shared__ float red[256];
    int tid = threadIdx.x;
    float mx = 0.f;
    for (int q = tid; q < Q; q += 256) {
        float d = sqrtf(fmaxf(minD2[q], 0.f));
        out[q] = d;
        mx = fmaxf(mx, d);
    }
    red[tid] = mx;
    __syncthreads();
    for (int s = 128; s; s >>= 1) {
        if (tid < s) red[tid] = fmaxf(red[tid], red[tid + s]);
        __syncthreads();
    }
    if (tid == 0) out[Q] = red[0];
}

// ---------- launch ----------
extern "C" void kernel_launch(void* const* d_in, const int* in_sizes, int n_in,
                              void* d_out, int out_size, void* d_ws, size_t ws_size,
                              hipStream_t stream) {
    const float* query = (const float*)d_in[0];
    const float* memv  = (const float*)d_in[1];
    float*       out   = (float*)d_out;

    const int Q = in_sizes[0] / DIMK;   // 4096
    const int M = in_sizes[1] / DIMK;   // 100000

    float* minD2 = (float*)d_ws;        // Q floats
    float* q2    = minD2 + Q;           // Q floats
    float* m2    = q2 + Q;              // M floats

    size_t base      = ((size_t)(2 * Q + M) * 4 + 63) & ~(size_t)63;
    size_t bf16bytes = (size_t)(Q + M) * DIMK * 2;
    bool   use_tdm   = ws_size >= base + bf16bytes + 64;

    init_min_kernel<<<(Q + 255) / 256, 256, 0, stream>>>(minD2, Q);
    sqnorm_kernel<<<(Q * 32 + 255) / 256, 256, 0, stream>>>(query, q2, Q);
    sqnorm_kernel<<<((size_t)M * 32 + 255) / 256, 256, 0, stream>>>(memv, m2, M);

    dim3 grid((Q + TQ - 1) / TQ, (M + TM - 1) / TM);   // q-major: memory tiles reused via L2

    if (use_tdm) {
        unsigned short* qbf = (unsigned short*)((char*)d_ws + base);
        unsigned short* mbf = qbf + (size_t)Q * DIMK;
        long long q4 = (long long)Q * DIMK / 4, m4 = (long long)M * DIMK / 4;
        cvt_bf16_kernel<<<4096, 256, 0, stream>>>(query, qbf, q4);
        cvt_bf16_kernel<<<16384, 256, 0, stream>>>(memv, mbf, m4);
        nn_gemm_tdm_kernel<<<grid, 256, 0, stream>>>(qbf, mbf, q2, m2, minD2, Q, M);
    } else {
        nn_gemm_kernel<<<grid, 256, 0, stream>>>(query, memv, q2, m2, minD2, Q, M);
    }

    finalize_kernel<<<1, 256, 0, stream>>>(minD2, out, Q);
}
